// Attention_70231305224528
// MI455X (gfx1250) — matrix operand order repaired and checked by reading, not compiled, not run on hardware
//
#include <hip/hip_runtime.h>
#include <hip/hip_bf16.h>

typedef __attribute__((ext_vector_type(16))) __bf16 v16bf;
typedef __attribute__((ext_vector_type(8)))  __bf16 v8bf;
typedef __attribute__((ext_vector_type(8)))  float  v8f;
typedef __attribute__((ext_vector_type(4)))  int    v4i;

#define BSZ  2048   // batch
#define QD   512    // Q_DIM (K of gemm1)
#define KD   1024   // K_DIM (N of gemm1/2, K of gemm2)
#define NS   16     // N_SAVED
#define HID  512
#define LAY  2

// ---- CDNA5 async global->LDS copy (ASYNCcnt path), with safe fallback ------
#if defined(__AMDGCN__) && __has_builtin(__builtin_amdgcn_global_load_async_to_lds_b128)
#define USE_ASYNC_LDS 1
#else
#define USE_ASYNC_LDS 0
#endif

#if USE_ASYNC_LDS
typedef __attribute__((address_space(1))) v4i* gv4i_p;  // global v4i*
typedef __attribute__((address_space(3))) v4i* lv4i_p;  // LDS v4i*

__device__ __forceinline__ void async_cp16(void* lds, const void* gmem) {
  __builtin_amdgcn_global_load_async_to_lds_b128(
      (gv4i_p)(gmem), (lv4i_p)(lds), 0, 0);
}
__device__ __forceinline__ void async_wait0() {
#if __has_builtin(__builtin_amdgcn_s_wait_asynccnt)
  __builtin_amdgcn_s_wait_asynccnt(0);
#else
  asm volatile("s_wait_asynccnt 0x0" ::: "memory");
#endif
}
#endif

// ---------------------------------------------------------------------------
// Kernel 1: Qbf[b,j] = bf16( sum_k x[b,k]*Wq[j,k] + bq[j] )
// A = x [2048 x 512] row-major, B^T = Wq [1024 x 512] row-major (NT GEMM).
// Block tile 64(M) x 128(N), 8 waves, each wave 32x32 via 4 WMMA accs.
// ---------------------------------------------------------------------------
__global__ __launch_bounds__(256) void gemm_q(const float* __restrict__ X,
                                              const float* __restrict__ Wq,
                                              const float* __restrict__ bq,
                                              __bf16* __restrict__ Qbf) {
  __shared__ __bf16 sA[64][40];    // [m][k], padded to 40 (80B rows)
  __shared__ __bf16 sB[128][40];   // [n][k], padded
  const int tid  = threadIdx.x;
  const int lane = tid & 31;
  const int wave = tid >> 5;
  const int wm = wave & 1;         // 0..1 (M)
  const int wn = wave >> 1;        // 0..3 (N)
  const int gm0 = blockIdx.y * 64;
  const int gn0 = blockIdx.x * 128;

  v8f acc[2][2] = {};

  for (int k0 = 0; k0 < QD; k0 += 32) {
    // stage A tile (fp32 -> bf16): 64x32; contiguous along k -> compiler packs
#pragma unroll
    for (int i = 0; i < 2; ++i) {
      int q = tid + i * 256;           // 0..511 float4s
      int r = q >> 3;
      int c = (q & 7) << 2;
      float4 f = *(const float4*)&X[(size_t)(gm0 + r) * QD + k0 + c];
      sA[r][c + 0] = (__bf16)f.x; sA[r][c + 1] = (__bf16)f.y;
      sA[r][c + 2] = (__bf16)f.z; sA[r][c + 3] = (__bf16)f.w;
    }
    // stage B tile (rows of Wq, contiguous along K): 128x32
#pragma unroll
    for (int i = 0; i < 4; ++i) {
      int q = tid + i * 256;           // 0..1023 float4s
      int r = q >> 3;
      int c = (q & 7) << 2;
      float4 f = *(const float4*)&Wq[(size_t)(gn0 + r) * QD + k0 + c];
      sB[r][c + 0] = (__bf16)f.x; sB[r][c + 1] = (__bf16)f.y;
      sB[r][c + 2] = (__bf16)f.z; sB[r][c + 3] = (__bf16)f.w;
    }
    __syncthreads();

    const int r16   = lane & 15;
    const int khalf = lane >> 4;
    v16bf af[2], bfr[2];
#pragma unroll
    for (int mi = 0; mi < 2; ++mi) {
      const __bf16* pa = &sA[wm * 32 + mi * 16 + r16][0];
      *((v8bf*)&af[mi] + 0) = *(const v8bf*)&pa[khalf * 8];
      *((v8bf*)&af[mi] + 1) = *(const v8bf*)&pa[16 + khalf * 8];
    }
#pragma unroll
    for (int ni = 0; ni < 2; ++ni) {
      const __bf16* pb = &sB[wn * 32 + ni * 16 + r16][0];
      *((v8bf*)&bfr[ni] + 0) = *(const v8bf*)&pb[khalf * 8];
      *((v8bf*)&bfr[ni] + 1) = *(const v8bf*)&pb[16 + khalf * 8];
    }
#pragma unroll
    for (int mi = 0; mi < 2; ++mi)
#pragma unroll
      for (int ni = 0; ni < 2; ++ni)
        acc[mi][ni] = __builtin_amdgcn_wmma_f32_16x16x32_bf16(
            false, af[mi], false, bfr[ni], (short)0, acc[mi][ni], false, false);
    __syncthreads();
  }

  const int col16 = lane & 15;
  const int rbase = (lane >> 4) * 8;
#pragma unroll
  for (int mi = 0; mi < 2; ++mi)
#pragma unroll
    for (int ni = 0; ni < 2; ++ni) {
      int gn = gn0 + wn * 32 + ni * 16 + col16;
      float bias = bq[gn];
#pragma unroll
      for (int j = 0; j < 8; ++j) {
        int gm = gm0 + wm * 32 + mi * 16 + rbase + j;
        Qbf[(size_t)gm * KD + gn] = (__bf16)(acc[mi][ni][j] + bias);
      }
    }
}

// ---------------------------------------------------------------------------
// Kernel 2: Qp[b,c] = sum_j Qbf[b,j] * Wk[j,c]   (NN GEMM, B transposed in LDS)
// A tile is a pure bf16 copy -> async global->LDS; B tile converts fp32->bf16
// with pairs packed along K so LDS writes are 4-byte ds_store_b32.
// ---------------------------------------------------------------------------
__global__ __launch_bounds__(256) void gemm_qp(const __bf16* __restrict__ Qbf,
                                               const float* __restrict__ Wk,
                                               float* __restrict__ Qp) {
  __shared__ __bf16 sA[64][40];
  __shared__ __bf16 sB[128][40];
  const int tid  = threadIdx.x;
  const int lane = tid & 31;
  const int wave = tid >> 5;
  const int wm = wave & 1;
  const int wn = wave >> 1;
  const int gm0 = blockIdx.y * 64;
  const int gn0 = blockIdx.x * 128;

  v8f acc[2][2] = {};

  const int ar  = tid >> 2;          // 0..63
  const int ach = (tid & 3) * 8;     // 0,8,16,24

  for (int k0 = 0; k0 < KD; k0 += 32) {
    // stage A tile (already bf16): one 16B chunk per thread
#if USE_ASYNC_LDS
    async_cp16(&sA[ar][ach], &Qbf[(size_t)(gm0 + ar) * KD + k0 + ach]);
#else
    *(v8bf*)&sA[ar][ach] = *(const v8bf*)&Qbf[(size_t)(gm0 + ar) * KD + k0 + ach];
#endif
    // stage B tile with transpose: sB[n][k] = bf16(Wk[k0+k][gn0+n]).
    // Two consecutive k rows per thread -> packed 4-byte LDS stores.
#pragma unroll
    for (int i = 0; i < 2; ++i) {
      int q  = tid + i * 256;        // 0..511
      int kk = (q >> 5) << 1;        // even k: 0,2,...,30
      int n  = (q & 31) << 2;        // 0..124
      float4 f0 = *(const float4*)&Wk[(size_t)(k0 + kk) * KD + gn0 + n];
      float4 f1 = *(const float4*)&Wk[(size_t)(k0 + kk + 1) * KD + gn0 + n];
      union { unsigned u; __bf16 h[2]; } p;
      p.h[0] = (__bf16)f0.x; p.h[1] = (__bf16)f1.x; *(unsigned*)&sB[n + 0][kk] = p.u;
      p.h[0] = (__bf16)f0.y; p.h[1] = (__bf16)f1.y; *(unsigned*)&sB[n + 1][kk] = p.u;
      p.h[0] = (__bf16)f0.z; p.h[1] = (__bf16)f1.z; *(unsigned*)&sB[n + 2][kk] = p.u;
      p.h[0] = (__bf16)f0.w; p.h[1] = (__bf16)f1.w; *(unsigned*)&sB[n + 3][kk] = p.u;
    }
#if USE_ASYNC_LDS
    async_wait0();
#endif
    __syncthreads();

    const int r16   = lane & 15;
    const int khalf = lane >> 4;
    v16bf af[2], bfr[2];
#pragma unroll
    for (int mi = 0; mi < 2; ++mi) {
      const __bf16* pa = &sA[wm * 32 + mi * 16 + r16][0];
      *((v8bf*)&af[mi] + 0) = *(const v8bf*)&pa[khalf * 8];
      *((v8bf*)&af[mi] + 1) = *(const v8bf*)&pa[16 + khalf * 8];
    }
#pragma unroll
    for (int ni = 0; ni < 2; ++ni) {
      const __bf16* pb = &sB[wn * 32 + ni * 16 + r16][0];
      *((v8bf*)&bfr[ni] + 0) = *(const v8bf*)&pb[khalf * 8];
      *((v8bf*)&bfr[ni] + 1) = *(const v8bf*)&pb[16 + khalf * 8];
    }
#pragma unroll
    for (int mi = 0; mi < 2; ++mi)
#pragma unroll
      for (int ni = 0; ni < 2; ++ni)
        acc[mi][ni] = __builtin_amdgcn_wmma_f32_16x16x32_bf16(
            false, af[mi], false, bfr[ni], (short)0, acc[mi][ni], false, false);
    __syncthreads();
  }

  const int col16 = lane & 15;
  const int rbase = (lane >> 4) * 8;
#pragma unroll
  for (int mi = 0; mi < 2; ++mi)
#pragma unroll
    for (int ni = 0; ni < 2; ++ni) {
      int gn = gn0 + wn * 32 + ni * 16 + col16;
#pragma unroll
      for (int j = 0; j < 8; ++j) {
        int gm = gm0 + wm * 32 + mi * 16 + rbase + j;
        Qp[(size_t)gm * KD + gn] = acc[mi][ni][j];
      }
    }
}

// ---------------------------------------------------------------------------
// Kernel 3: per-b scores/softmax/argmax -> pos[b]
// scores[b,n] = feat(b,n)·Qp[b];  V[b,n] = feat(b,n)·Wv + bv
// feat(b,n) = concat(h[n,1,b,:], c[n,1,b,:]).  Memory-bound: streams 134 MB.
// ---------------------------------------------------------------------------
__global__ __launch_bounds__(256) void select_pos(const float* __restrict__ Hs,
                                                  const float* __restrict__ Cs,
                                                  const float* __restrict__ Qp,
                                                  const float* __restrict__ Wv,
                                                  const float* __restrict__ bv,
                                                  int* __restrict__ pos) {
  __shared__ float sQ[KD];
  __shared__ float sW[KD];
  __shared__ float sS[NS];
  __shared__ float sV[NS];
  const int b   = blockIdx.x;
  const int tid = threadIdx.x;

  // 256 threads x one float4 each for sQ and sW (pure copies -> async path)
#if USE_ASYNC_LDS
  async_cp16(&((float4*)sQ)[tid], &((const float4*)(Qp + (size_t)b * KD))[tid]);
  async_cp16(&((float4*)sW)[tid], &((const float4*)Wv)[tid]);
  async_wait0();
#else
  ((float4*)sQ)[tid] = ((const float4*)(Qp + (size_t)b * KD))[tid];
  ((float4*)sW)[tid] = ((const float4*)Wv)[tid];
#endif
  __syncthreads();

  const int lane = tid & 31;
  const int wave = tid >> 5;
#pragma unroll
  for (int nn = 0; nn < 2; ++nn) {
    const int n = wave * 2 + nn;
    const float* __restrict__ ph = Hs + ((size_t)(n * LAY + (LAY - 1)) * BSZ + b) * HID;
    const float* __restrict__ pc = Cs + ((size_t)(n * LAY + (LAY - 1)) * BSZ + b) * HID;
    float as = 0.f, av = 0.f;
#pragma unroll 4
    for (int i = 0; i < HID / 32; ++i) {
      int k = lane + 32 * i;
      float f = ph[k];
      as += f * sQ[k];
      av += f * sW[k];
    }
#pragma unroll 4
    for (int i = 0; i < HID / 32; ++i) {
      int k = lane + 32 * i;
      float f = pc[k];
      as += f * sQ[HID + k];
      av += f * sW[HID + k];
    }
#pragma unroll
    for (int off = 16; off; off >>= 1) {
      as += __shfl_xor(as, off, 32);
      av += __shfl_xor(av, off, 32);
    }
    if (lane == 0) { sS[n] = as; sV[n] = av + bv[0]; }
  }
  __syncthreads();

  if (tid == 0) {
    float m = sS[0];
#pragma unroll
    for (int n = 1; n < NS; ++n) m = fmaxf(m, sS[n]);
    int best = 0;
    float bestv = -3.402823466e38f;
#pragma unroll
    for (int n = 0; n < NS; ++n) {
      // softmax(s)*V has argmax == argmax of exp(s-m)*V (Z > 0 common factor)
      float ctx = __expf(sS[n] - m) * sV[n];
      if (ctx > bestv) { bestv = ctx; best = n; }
    }
    pos[b] = best;
  }
}

// ---------------------------------------------------------------------------
// Kernel 4: gather selected states.
// out = [out_h (L,B,H), out_c (L,B,H)], out_h[l,b,:] = Hs[pos[b], l, b, :]
// ---------------------------------------------------------------------------
__global__ __launch_bounds__(256) void gather_out(const float* __restrict__ Hs,
                                                  const float* __restrict__ Cs,
                                                  const int* __restrict__ pos,
                                                  float* __restrict__ out) {
  const int b   = blockIdx.x;
  const int p   = pos[b];
  const int tid = threadIdx.x;
  float* outh = out;
  float* outc = out + (size_t)LAY * BSZ * HID;
#pragma unroll
  for (int i = 0; i < 2; ++i) {
    int q   = tid + i * 256;     // 0..511 float4s (4 rows x 128)
    int row = q >> 7;            // 0: h,l0  1: h,l1  2: c,l0  3: c,l1
    int e   = (q & 127) << 2;
    int l   = row & 1;
    const float* src = (row < 2 ? Hs : Cs) + ((size_t)(p * LAY + l) * BSZ + b) * HID + e;
    float* dst = (row < 2 ? outh : outc) + ((size_t)l * BSZ + b) * HID + e;
    *(float4*)dst = *(const float4*)src;
  }
}

// ---------------------------------------------------------------------------
extern "C" void kernel_launch(void* const* d_in, const int* in_sizes, int n_in,
                              void* d_out, int out_size, void* d_ws, size_t ws_size,
                              hipStream_t stream) {
  const float* x  = (const float*)d_in[0];
  const float* Hs = (const float*)d_in[1];
  const float* Cs = (const float*)d_in[2];
  const float* Wq = (const float*)d_in[3];
  const float* bq = (const float*)d_in[4];
  const float* Wk = (const float*)d_in[5];
  // d_in[6] (bk): adds a constant over n to scores -> softmax/argmax invariant, unused.
  const float* Wv = (const float*)d_in[7];
  const float* bv = (const float*)d_in[8];
  float* out = (float*)d_out;

  char* ws = (char*)d_ws;
  __bf16* Qbf = (__bf16*)ws;                                        // 4 MiB
  float*  Qp  = (float*)(ws + (size_t)BSZ * KD * sizeof(__bf16));   // 8 MiB
  int*    pos = (int*)(ws + (size_t)BSZ * KD * sizeof(__bf16)
                          + (size_t)BSZ * KD * sizeof(float));      // 8 KiB

  dim3 gg(KD / 128, BSZ / 64);  // (8, 32)
  gemm_q<<<gg, 256, 0, stream>>>(x, Wq, bq, Qbf);
  gemm_qp<<<gg, 256, 0, stream>>>(Qbf, Wk, Qp);
  select_pos<<<BSZ, 256, 0, stream>>>(Hs, Cs, Qp, Wv, bv, pos);
  gather_out<<<BSZ, 256, 0, stream>>>(Hs, Cs, pos, out);
}
